// LSTMWithGaussianAttention_3676492005743
// MI455X (gfx1250) — compile-verified, matrix operand order checked by reading, not run on hardware
//
#include <hip/hip_runtime.h>
#include <hip/hip_bf16.h>

// Problem dims (fixed by the reference)
#define B_   128
#define T_   800
#define H_   512
#define K_   10
#define C_   80
#define U_   64
#define IN_  3
#define G4H  2048            // 4*H
#define KCAT 640             // padded K: [0..2]=x, [3..82]=window, [83..594]=h, rest 0
#define KTILES 20            // KCAT/32
#define NTILES 128           // G4H/16
#define GRID 128             // persistent blocks (also = B_)
#define GEMM_BLOCKS 64       // blocks carrying the GEMM phase (2 N-tiles each)

typedef _Float16 v16h __attribute__((ext_vector_type(16)));
typedef _Float16 v8h  __attribute__((ext_vector_type(8)));
typedef float    v8f  __attribute__((ext_vector_type(8)));

// ---------------------------------------------------------------------------
// Prep 0: bias = b_ih + b_hh
// ---------------------------------------------------------------------------
__global__ void prep_bias(const float* __restrict__ b_ih,
                          const float* __restrict__ b_hh,
                          float* __restrict__ bias) {
  int i = blockIdx.x * blockDim.x + threadIdx.x;
  if (i < G4H) bias[i] = b_ih[i] + b_hh[i];
}

// ---------------------------------------------------------------------------
// Prep 1: pack Wcat = [W_ih | W_hh | 0] (f32) into the per-lane WMMA B-fragment
// layout (f16).  B tile is 32(K) x 16(N); lane L holds 16 halves:
//   e in [0,16): v = e>>1 ;  k = 2*(v&3) + (e&1) + ((L>=16)?8:0) + (v>=4?16:0)
//   n = (L&15)
// Storage: Bpack[((tN*KTILES + tK)*32 + L)*16 + e]  -> flat id order.
// ---------------------------------------------------------------------------
__global__ void prep_pack(const float* __restrict__ W_ih,
                          const float* __restrict__ W_hh,
                          _Float16* __restrict__ Bpack) {
  int id = blockIdx.x * blockDim.x + threadIdx.x;        // 0 .. 1310719
  if (id >= NTILES * KTILES * 32 * 16) return;
  int e    = id & 15;
  int L    = (id >> 4) & 31;
  int tile = id >> 9;
  int tK   = tile % KTILES;
  int tN   = tile / KTILES;
  int v    = e >> 1;
  int k    = tK * 32 + 2 * (v & 3) + (e & 1) + ((L >> 4) & 1) * 8 + (v >> 2) * 16;
  int n    = tN * 16 + (L & 15);
  float val;
  if (k < IN_ + C_)           val = W_ih[n * (IN_ + C_) + k];
  else if (k < IN_ + C_ + H_) val = W_hh[n * H_ + (k - IN_ - C_)];
  else                        val = 0.0f;
  Bpack[id] = (_Float16)val;
}

// ---------------------------------------------------------------------------
// Prep 2: init state — Acat gets x_0 and zeros, c=0, kappa=0, barrier=0
// ---------------------------------------------------------------------------
__global__ void prep_state(const float* __restrict__ strokes,
                           _Float16* __restrict__ Acat,
                           float* __restrict__ c_state,
                           float* __restrict__ kappa,
                           unsigned* __restrict__ bar) {
  int id = blockIdx.x * blockDim.x + threadIdx.x;
  if (id < B_ * KCAT) {
    int b = id / KCAT, k = id % KCAT;
    _Float16 val = (_Float16)0.0f;
    if (k < IN_) val = (_Float16)strokes[((size_t)b * T_ + 0) * IN_ + k];
    Acat[id] = val;
  }
  if (id < B_ * H_)  c_state[id] = 0.0f;
  if (id < B_ * K_)  kappa[id]   = 0.0f;
  if (id == 0)       *bar        = 0u;
}

// ---------------------------------------------------------------------------
// Device-wide sense-free barrier (monotonic counter, agent scope).
// All blocks call it the same number of times.
// ---------------------------------------------------------------------------
__device__ __forceinline__ void grid_barrier(unsigned* bar, unsigned target) {
  __threadfence();                 // release this thread's global writes
  __syncthreads();                 // whole block arrived
  if (threadIdx.x == 0) {
    __hip_atomic_fetch_add(bar, 1u, __ATOMIC_ACQ_REL, __HIP_MEMORY_SCOPE_AGENT);
    while (__hip_atomic_load(bar, __ATOMIC_ACQUIRE, __HIP_MEMORY_SCOPE_AGENT) < target)
      __builtin_amdgcn_s_sleep(1);
  }
  __syncthreads();
  __threadfence();                 // acquire side: invalidate stale lines
}

// ---------------------------------------------------------------------------
// Persistent kernel: all 800 timesteps in one launch.
//   Phase A (blocks 0..63): gates[128,2048] = Acat[128,640] @ Wcat.T
//     Packed weights for this block's 32 N-columns live in LDS for the whole
//     sequence. Each wave: 16M x 32N (one A fragment feeds two WMMAs).
//   Phase B (all 128 blocks): LSTM cell + window projection + attention for
//     batch row b = blockIdx.x; stages h/window/x_{t+1} (f16) for next step.
// ---------------------------------------------------------------------------
__global__ __launch_bounds__(256)
void handwriting_persistent(const _Float16* __restrict__ Bpack,
                            _Float16* Acat,
                            float* gates,
                            const float* __restrict__ bias,
                            float* c_state,
                            float* kappa_state,
                            const float* __restrict__ strokes,
                            const int* __restrict__ sentences,
                            const float* __restrict__ mask,
                            const float* __restrict__ W_win,
                            const float* __restrict__ b_win,
                            float* __restrict__ out_h,
                            float* __restrict__ out_w,
                            float* __restrict__ out_phi,
                            unsigned* bar) {
  __shared__ __align__(32) _Float16 Blds[2 * KTILES * 512];   // 40 KB, resident
  __shared__ float h_lds[H_];
  __shared__ float p_lds[32];
  __shared__ float al[K_], be[K_], ka[K_];
  __shared__ float phi_lds[U_];

  const int tid  = threadIdx.x;
  const int wave = tid >> 5;
  const int lane = tid & 31;
  const int blk  = blockIdx.x;

  // ---- one-time: stage packed weights for this block's two N-tiles ----
  if (blk < GEMM_BLOCKS) {
    const uint4* s = (const uint4*)(Bpack + (size_t)(2 * blk) * (KTILES * 512));
    uint4*       d = (uint4*)Blds;
#pragma unroll
    for (int i = 0; i < 10; ++i) d[tid + 256 * i] = s[tid + 256 * i];
  }
  __syncthreads();

  // per-wave constant GEMM addressing
  const int M0   = wave * 16;
  const int m    = M0 + (lane & 15);
  const int koff = ((lane >> 4) & 1) * 8;             // hi half-wave takes K+8
  const int row0 = M0 + ((lane >> 4) & 1) * 8;        // C/D row base
  const int col0 = (2 * blk) * 16 + (lane & 15);      // first N tile column

  unsigned bt = 0;
#pragma unroll 1
  for (int t = 0; t < T_; ++t) {
    // ================= Phase A: gate GEMM =================
    if (blk < GEMM_BLOCKS) {
      const _Float16* arow = Acat + m * KCAT + koff;
      v8f acc0 = {}, acc1 = {};
#pragma unroll 5
      for (int tK = 0; tK < KTILES; ++tK) {
        const int K0 = tK * 32;
        v8h a0 = *(const v8h*)(arow + K0);            // K0+koff   .. +7
        v8h a1 = *(const v8h*)(arow + K0 + 16);       // K0+koff+16.. +23
        v16h a = __builtin_shufflevector(a0, a1, 0, 1, 2, 3, 4, 5, 6, 7,
                                                 8, 9, 10, 11, 12, 13, 14, 15);
        v16h b0 = *(const v16h*)(Blds + tK * 512 + lane * 16);
        v16h b1 = *(const v16h*)(Blds + KTILES * 512 + tK * 512 + lane * 16);
        acc0 = __builtin_amdgcn_wmma_f32_16x16x32_f16(false, a, false, b0,
                                                      (short)0, acc0, false, false);
        acc1 = __builtin_amdgcn_wmma_f32_16x16x32_f16(false, a, false, b1,
                                                      (short)0, acc1, false, false);
      }
#pragma unroll
      for (int r = 0; r < 8; ++r) {
        gates[(size_t)(row0 + r) * G4H + col0]      = acc0[r];
        gates[(size_t)(row0 + r) * G4H + col0 + 16] = acc1[r];
      }
    }
    bt += gridDim.x;
    grid_barrier(bar, bt);

    // ================= Phase B: cell + attention (batch row = blk) ========
    {
      const int b = blk;
      const float* g = gates + (size_t)b * G4H;
#pragma unroll
      for (int s = 0; s < 2; ++s) {
        int j = tid + s * 256;
        float ii = g[j]        + bias[j];
        float ff = g[H_ + j]   + bias[H_ + j];
        float gg = g[2*H_ + j] + bias[2*H_ + j];
        float oo = g[3*H_ + j] + bias[3*H_ + j];
        float si = 1.0f / (1.0f + __expf(-ii));
        float sf = 1.0f / (1.0f + __expf(-ff));
        float so = 1.0f / (1.0f + __expf(-oo));
        float cn = sf * c_state[b * H_ + j] + si * tanhf(gg);
        float h  = so * tanhf(cn);
        c_state[b * H_ + j] = cn;
        h_lds[j] = h;
        out_h[((size_t)b * T_ + t) * H_ + j] = h;
        Acat[b * KCAT + (IN_ + C_) + j] = (_Float16)h;   // h for next GEMM
      }
      __syncthreads();

      // p = h @ W_win.T + b_win : 30 dots of length 512, one wave each
      for (int r = wave; r < 3 * K_; r += 8) {
        const float* wr = W_win + r * H_;
        float s = 0.0f;
        for (int k2 = lane; k2 < H_; k2 += 32) s += h_lds[k2] * wr[k2];
#pragma unroll
        for (int off = 16; off; off >>= 1) s += __shfl_xor(s, off, 32);
        if (lane == 0) p_lds[r] = s + b_win[r];
      }
      __syncthreads();

      if (tid < K_) {
        al[tid] = __expf(p_lds[tid]);
        be[tid] = __expf(p_lds[K_ + tid]);
        float kk = kappa_state[b * K_ + tid] + __expf(p_lds[2 * K_ + tid]);
        kappa_state[b * K_ + tid] = kk;
        ka[tid] = kk;
      }
      __syncthreads();

      if (tid < U_) {
        float u = (float)tid, s = 0.0f;
#pragma unroll
        for (int k2 = 0; k2 < K_; ++k2) {
          float d = ka[k2] - u;
          s += al[k2] * __expf(-be[k2] * d * d);
        }
        phi_lds[tid] = s;
        out_phi[((size_t)b * T_ + t) * U_ + tid] = s;
      }
      __syncthreads();

      if (tid < C_) {
        float s = 0.0f;
        for (int u = 0; u < U_; ++u)
          if (sentences[b * U_ + u] == tid) s += phi_lds[u] * mask[b * U_ + u];
        out_w[((size_t)b * T_ + t) * C_ + tid] = s;
        Acat[b * KCAT + IN_ + tid] = (_Float16)s;        // window for next GEMM
      }

      if (tid < IN_ && (t + 1) < T_)
        Acat[b * KCAT + tid] =
            (_Float16)strokes[((size_t)b * T_ + (t + 1)) * IN_ + tid];
      __syncthreads();   // protect h_lds/phi_lds reuse next iteration
    }
    bt += gridDim.x;
    grid_barrier(bar, bt);
  }
}

// ---------------------------------------------------------------------------
extern "C" void kernel_launch(void* const* d_in, const int* in_sizes, int n_in,
                              void* d_out, int out_size, void* d_ws, size_t ws_size,
                              hipStream_t stream) {
  const float* strokes   = (const float*)d_in[0];
  const int*   sentences = (const int*)  d_in[1];
  const float* smask     = (const float*)d_in[2];
  const float* W_ih      = (const float*)d_in[3];
  const float* W_hh      = (const float*)d_in[4];
  const float* b_ih      = (const float*)d_in[5];
  const float* b_hh      = (const float*)d_in[6];
  const float* W_win     = (const float*)d_in[7];
  const float* b_win     = (const float*)d_in[8];

  float* out_h   = (float*)d_out;
  float* out_w   = out_h + (size_t)B_ * T_ * H_;
  float* out_phi = out_w + (size_t)B_ * T_ * C_;

  char* ws = (char*)d_ws;
  _Float16* Acat  = (_Float16*)ws; ws += (size_t)B_ * KCAT * 2;          // 160 KB
  _Float16* Bpack = (_Float16*)ws; ws += (size_t)G4H * KCAT * 2;         // 2.5 MB
  float* gates    = (float*)ws;    ws += (size_t)B_ * G4H * 4;           // 1 MB
  float* c_state  = (float*)ws;    ws += (size_t)B_ * H_ * 4;            // 256 KB
  float* kappa    = (float*)ws;    ws += (size_t)B_ * K_ * 4;
  float* bias     = (float*)ws;    ws += (size_t)G4H * 4;
  unsigned* bar   = (unsigned*)ws; ws += 256;

  prep_bias <<<(G4H + 255) / 256, 256, 0, stream>>>(b_ih, b_hh, bias);
  prep_pack <<<(NTILES * KTILES * 32 * 16 + 255) / 256, 256, 0, stream>>>(W_ih, W_hh, Bpack);
  prep_state<<<(B_ * KCAT + 255) / 256, 256, 0, stream>>>(strokes, Acat, c_state, kappa, bar);

  handwriting_persistent<<<GRID, 256, 0, stream>>>(
      Bpack, Acat, gates, bias, c_state, kappa,
      strokes, sentences, smask, W_win, b_win,
      out_h, out_w, out_phi, bar);
}
